// GIN_pyg_80255758893329
// MI455X (gfx1250) — matrix operand order repaired and checked by reading, not compiled, not run on hardware
//
#include <hip/hip_runtime.h>
#include <hip/hip_bf16.h>

#define N_NODES   20000
#define N_EDGES   640000
#define CH        128
#define NUM_GRAPHS 64

typedef __attribute__((ext_vector_type(16))) __bf16 bf16x16;
typedef __attribute__((ext_vector_type(8)))  float  floatx8;

union FragB16 {
    bf16x16 v;
    uint4   q[2];
};

static __device__ __forceinline__ unsigned short f32_to_bf16_rne(float f) {
    union { float f; unsigned int u; } cv;
    cv.f = f;
    unsigned int lsb = (cv.u >> 16) & 1u;
    unsigned int r   = cv.u + 0x7fffu + lsb;
    return (unsigned short)(r >> 16);
}

// ---------------------------------------------------------------------------
// Zero a float buffer.
__global__ void zero_kernel(float* __restrict__ p, int n) {
    int i = blockIdx.x * blockDim.x + threadIdx.x;
    if (i < n) p[i] = 0.0f;
}

// ---------------------------------------------------------------------------
// Swizzle 4 weight matrices (128x128 f32, row-major K x N) into WMMA B-fragment
// order, bf16. Layout per matrix: [n_tile(8)][kk(4)][lane(32)][elem(16)].
// lane<16 : K = 32*kk + e,  N = 16*n_tile + lane
// lane>=16: K = 32*kk + 16 + e, N = 16*n_tile + (lane-16)
__global__ void wswz_kernel(const float* __restrict__ W0, const float* __restrict__ W1,
                            const float* __restrict__ W2, const float* __restrict__ W3,
                            unsigned short* __restrict__ out) {
    int i = blockIdx.x * blockDim.x + threadIdx.x;
    if (i >= 4 * 128 * 128) return;
    int m = i >> 14;
    int j = i & 16383;
    const float* W = (m == 0) ? W0 : (m == 1) ? W1 : (m == 2) ? W2 : W3;
    int e    = j & 15;
    int lane = (j >> 4) & 31;
    int kk   = (j >> 9) & 3;
    int nt   = (j >> 11) & 7;
    int k = 32 * kk + ((lane >= 16) ? 16 : 0) + e;
    int n = 16 * nt + (lane & 15);
    out[i] = f32_to_bf16_rne(W[k * CH + n]);
}

// ---------------------------------------------------------------------------
// Edge scatter-add: one thread = one edge x 4 channels.
__global__ void scatter_add_kernel(const float* __restrict__ x,
                                   const long long* __restrict__ src,
                                   const long long* __restrict__ dst,
                                   float* __restrict__ agg) {
    int t = blockIdx.x * blockDim.x + threadIdx.x;
    if (t >= N_EDGES * 32) return;
    int e = t >> 5;
    int c = (t & 31) << 2;
    long long s = src[e];
    long long d = dst[e];
    const float4 v = *(const float4*)(x + s * CH + c);
    float* o = agg + d * CH + c;
    atomicAdd(o + 0, v.x);
    atomicAdd(o + 1, v.y);
    atomicAdd(o + 2, v.z);
    atomicAdd(o + 3, v.w);
}

// ---------------------------------------------------------------------------
// h = x + agg, converted to bf16 (WMMA A operand, row-major).
__global__ void add_convert_kernel(const float* __restrict__ x,
                                   const float* __restrict__ agg,
                                   unsigned short* __restrict__ out, int n) {
    int i = blockIdx.x * blockDim.x + threadIdx.x;
    if (i < n) out[i] = f32_to_bf16_rne(x[i] + agg[i]);
}

// ---------------------------------------------------------------------------
// Fused GIN MLP: out = relu( relu(A @ W1 + b1) @ W2 + b2 )
// A: [M x 128] bf16 row-major. One block = 16 rows x 128 cols; 8 waves, each
// wave owns one 16-wide N tile. Mid activation staged in LDS as bf16.
__global__ __launch_bounds__(256) void gin_mlp_kernel(
    const unsigned short* __restrict__ A,
    const unsigned short* __restrict__ Wswz1, const float* __restrict__ b1,
    const unsigned short* __restrict__ Wswz2, const float* __restrict__ b2,
    float* __restrict__ out) {
    __shared__ __align__(16) unsigned short sA[16 * CH];
    __shared__ __align__(16) unsigned short sMid[16 * CH];

    const int tid  = threadIdx.x;
    const int wave = tid >> 5;            // n-tile index, 0..7
    const int lane = tid & 31;
    const int row0 = blockIdx.x * 16;     // M = 20000 -> 1250 exact tiles

    // Cooperative load of the 16x128 bf16 A tile (4 KB) into LDS.
    {
        const uint4* gsrc = (const uint4*)(A + (size_t)row0 * CH);
        uint4* ldst = (uint4*)sA;
        ldst[tid] = gsrc[tid];
    }
    __syncthreads();

    const int halfsel = (lane < 16) ? 0 : 16;  // in bf16 elements
    const int mrow    = lane & 15;
    const int ncol    = wave * 16 + (lane & 15);

    // ---- phase 1: C1 = A(16x128) @ W1(128x16) ----
    floatx8 acc = {};
#pragma unroll
    for (int kk = 0; kk < 4; ++kk) {
        FragB16 a, b;
        const unsigned short* ap = sA + mrow * CH + kk * 32 + halfsel;
        a.q[0] = *(const uint4*)(ap);        // K chunk +0..7
        a.q[1] = *(const uint4*)(ap + 8);    // K chunk +8..15
        const unsigned short* wp = Wswz1 + (size_t)((wave * 4 + kk) * 32 + lane) * 16;
        b.q[0] = *(const uint4*)(wp);
        b.q[1] = *(const uint4*)(wp + 8);
        acc = __builtin_amdgcn_wmma_f32_16x16x32_bf16(
            false, a.v, false, b.v, (short)0, acc, false, false);
    }

    // bias + relu, write bf16 mid tile to LDS
    const float bias1 = b1[ncol];
#pragma unroll
    for (int v = 0; v < 8; ++v) {
        int m = v + ((lane < 16) ? 0 : 8);
        float r = acc[v] + bias1;
        r = r > 0.0f ? r : 0.0f;
        sMid[m * CH + ncol] = f32_to_bf16_rne(r);
    }
    __syncthreads();

    // ---- phase 2: C2 = Mid(16x128) @ W2(128x16) ----
    floatx8 acc2 = {};
#pragma unroll
    for (int kk = 0; kk < 4; ++kk) {
        FragB16 a, b;
        const unsigned short* ap = sMid + mrow * CH + kk * 32 + halfsel;
        a.q[0] = *(const uint4*)(ap);
        a.q[1] = *(const uint4*)(ap + 8);
        const unsigned short* wp = Wswz2 + (size_t)((wave * 4 + kk) * 32 + lane) * 16;
        b.q[0] = *(const uint4*)(wp);
        b.q[1] = *(const uint4*)(wp + 8);
        acc2 = __builtin_amdgcn_wmma_f32_16x16x32_bf16(
            false, a.v, false, b.v, (short)0, acc2, false, false);
    }

    const float bias2 = b2[ncol];
#pragma unroll
    for (int v = 0; v < 8; ++v) {
        int m = v + ((lane < 16) ? 0 : 8);
        float r = acc2[v] + bias2;
        r = r > 0.0f ? r : 0.0f;           // outer relu of each conv layer
        out[(size_t)(row0 + m) * CH + ncol] = r;
    }
}

// ---------------------------------------------------------------------------
// Segment-sum pooling: one thread = one node x 4 channels; lane for c==0 also
// bumps the per-graph count.
__global__ void pool_kernel(const float* __restrict__ h,
                            const long long* __restrict__ batch,
                            float* __restrict__ psum, float* __restrict__ pcnt) {
    int t = blockIdx.x * blockDim.x + threadIdx.x;
    if (t >= N_NODES * 32) return;
    int nd = t >> 5;
    int c  = (t & 31) << 2;
    long long g = batch[nd];
    const float4 v = *(const float4*)(h + (size_t)nd * CH + c);
    float* o = psum + g * CH + c;
    atomicAdd(o + 0, v.x);
    atomicAdd(o + 1, v.y);
    atomicAdd(o + 2, v.z);
    atomicAdd(o + 3, v.w);
    if ((t & 31) == 0) atomicAdd(&pcnt[g], 1.0f);
}

// ---------------------------------------------------------------------------
// Final FC: out[g] = (psum[g] . fcw) / max(cnt,1) + fcb
__global__ void final_kernel(const float* __restrict__ psum,
                             const float* __restrict__ pcnt,
                             const float* __restrict__ fcw,
                             const float* __restrict__ fcb,
                             float* __restrict__ out) {
    int g = blockIdx.x * blockDim.x + threadIdx.x;
    if (g >= NUM_GRAPHS) return;
    float cnt = pcnt[g];
    cnt = cnt > 1.0f ? cnt : 1.0f;
    float acc = 0.0f;
#pragma unroll 4
    for (int c = 0; c < CH; ++c) acc += psum[g * CH + c] * fcw[c];
    out[g] = acc / cnt + fcb[0];
}

// ---------------------------------------------------------------------------
extern "C" void kernel_launch(void* const* d_in, const int* in_sizes, int n_in,
                              void* d_out, int out_size, void* d_ws, size_t ws_size,
                              hipStream_t stream) {
    (void)in_sizes; (void)n_in; (void)out_size; (void)ws_size;

    const float*     x    = (const float*)d_in[0];
    const long long* ei   = (const long long*)d_in[1];   // [2 x N_EDGES]
    const long long* batc = (const long long*)d_in[2];
    const float*     W1a  = (const float*)d_in[3];
    const float*     b1a  = (const float*)d_in[4];
    const float*     W1b  = (const float*)d_in[5];
    const float*     b1b  = (const float*)d_in[6];
    const float*     W2a  = (const float*)d_in[7];
    const float*     b2a  = (const float*)d_in[8];
    const float*     W2b  = (const float*)d_in[9];
    const float*     b2b  = (const float*)d_in[10];
    const float*     fcw  = (const float*)d_in[11];
    const float*     fcb  = (const float*)d_in[12];
    float* out = (float*)d_out;

    const long long* esrc = ei;
    const long long* edst = ei + N_EDGES;

    // Workspace layout (bytes, all 16B aligned)
    char* ws = (char*)d_ws;
    float*          agg   = (float*)(ws + 0);                       // 10,240,000 B
    unsigned short* Abf   = (unsigned short*)(ws + 10240000);       //  5,120,000 B
    float*          hbuf  = (float*)(ws + 15360000);                // 10,240,000 B
    unsigned short* Wswz  = (unsigned short*)(ws + 25600000);       //    131,072 B
    float*          psum  = (float*)(ws + 25731072);                //     32,768 B
    float*          pcnt  = (float*)(ws + 25763840);                //        256 B

    unsigned short* Wz1a = Wswz + 0 * 16384;
    unsigned short* Wz1b = Wswz + 1 * 16384;
    unsigned short* Wz2a = Wswz + 2 * 16384;
    unsigned short* Wz2b = Wswz + 3 * 16384;

    const int feat_n   = N_NODES * CH;                 // 2,560,000
    const int edge_thr = N_EDGES * 32;                 // 20,480,000
    const int node_thr = N_NODES * 32;                 //    640,000
    const int TB = 256;

    // 0) swizzle all weights to bf16 WMMA fragments
    wswz_kernel<<<(4 * 16384 + TB - 1) / TB, TB, 0, stream>>>(W1a, W1b, W2a, W2b, Wswz);

    // ---- conv 1 ----
    zero_kernel<<<(feat_n + TB - 1) / TB, TB, 0, stream>>>(agg, feat_n);
    scatter_add_kernel<<<(edge_thr + TB - 1) / TB, TB, 0, stream>>>(x, esrc, edst, agg);
    add_convert_kernel<<<(feat_n + TB - 1) / TB, TB, 0, stream>>>(x, agg, Abf, feat_n);
    gin_mlp_kernel<<<N_NODES / 16, 256, 0, stream>>>(Abf, Wz1a, b1a, Wz1b, b1b, hbuf);

    // ---- conv 2 ----
    zero_kernel<<<(feat_n + TB - 1) / TB, TB, 0, stream>>>(agg, feat_n);
    scatter_add_kernel<<<(edge_thr + TB - 1) / TB, TB, 0, stream>>>(hbuf, esrc, edst, agg);
    add_convert_kernel<<<(feat_n + TB - 1) / TB, TB, 0, stream>>>(hbuf, agg, Abf, feat_n);
    gin_mlp_kernel<<<N_NODES / 16, 256, 0, stream>>>(Abf, Wz2a, b2a, Wz2b, b2b, hbuf);

    // ---- pooling + FC ----
    zero_kernel<<<(NUM_GRAPHS * CH + NUM_GRAPHS + TB - 1) / TB, TB, 0, stream>>>(
        psum, NUM_GRAPHS * CH + NUM_GRAPHS);  // psum and pcnt are contiguous
    pool_kernel<<<(node_thr + TB - 1) / TB, TB, 0, stream>>>(hbuf, batc, psum, pcnt);
    final_kernel<<<1, 64, 0, stream>>>(psum, pcnt, fcw, fcb, out);
}
